// CausalSelfAttention_45234595561646
// MI455X (gfx1250) — matrix operand order repaired and checked by reading, not compiled, not run on hardware
//
#include <hip/hip_runtime.h>
#include <hip/hip_bf16.h>

typedef __attribute__((ext_vector_type(16))) __bf16 v16bf;
typedef __attribute__((ext_vector_type(8)))  __bf16 v8bf;
typedef __attribute__((ext_vector_type(8)))  float  v8f;
typedef __attribute__((ext_vector_type(4)))  unsigned int u32x4;
typedef __attribute__((ext_vector_type(8)))  int i32x8;
typedef __attribute__((ext_vector_type(4)))  int i32x4;

#if __has_builtin(__builtin_amdgcn_exp2f)
#define EXP2F __builtin_amdgcn_exp2f
#else
#define EXP2F exp2f
#endif

#if __has_builtin(__builtin_amdgcn_tensor_load_to_lds) && \
    __has_builtin(__builtin_amdgcn_s_wait_tensorcnt)
#define USE_TDM 1
#endif

static __device__ inline __bf16 f2bf(float f) {
  unsigned u = __float_as_uint(f);
  unsigned r = (u + 0x7FFFu + ((u >> 16) & 1u)) >> 16;
  unsigned short h = (unsigned short)r;
  return __builtin_bit_cast(__bf16, h);
}

// A-fragment for v_wmma_f32_16x16x32_bf16: elements 0..7 = K[c0..c0+7],
// elements 8..15 = K[c0+16..c0+23]; caller passes the two 16B chunk pointers.
static __device__ inline v16bf ld_a16(const __bf16* lo, const __bf16* hi) {
  v8bf l = *(const v8bf*)lo;
  v8bf h = *(const v8bf*)hi;
  v16bf r;
#pragma unroll
  for (int i = 0; i < 8; ++i) { r[i] = l[i]; r[i + 8] = h[i]; }
  return r;
}

static __device__ inline v8f wmma_bf16(v16bf a, v16bf b, v8f c) {
  return __builtin_amdgcn_wmma_f32_16x16x32_bf16(
      /*neg_a=*/false, a, /*neg_b=*/false, b,
      /*c_mod=*/(short)0, c, /*reuse_a=*/false, /*reuse_b=*/false);
}

#ifdef USE_TDM
// TDM: DMA a 2D tile (tile_x elems wide, tile_y rows, 2-byte elems, row
// stride stride_x elems) from global into LDS at byte offset lds_off.
// D# layout per CDNA5 ISA 8.3/8.4: group0 = {count=1, lds_addr,
// global_addr[56:0], type=2}; group1 = {mask=0, data_size=1(2B), big tensor
// dims (no OOB), tile dims, dim0 stride}; groups 2/3 zero (<=2D tensor).
static __device__ inline void tdm_load_2d(unsigned lds_off, const void* gaddr,
                                          unsigned tile_x, unsigned tile_y,
                                          unsigned stride_x) {
  unsigned long long ga = (unsigned long long)(uintptr_t)gaddr;
  u32x4 g0;
  g0[0] = 1u;                                        // count=1 (valid), user D#
  g0[1] = lds_off;                                   // LDS byte address
  g0[2] = (unsigned)ga;                              // global_addr[31:0]
  g0[3] = (unsigned)((ga >> 32) & 0x01FFFFFFu) | (2u << 30);  // addr[56:32]|type=2
  i32x8 g1;
  g1[0] = (int)(1u << 16);                           // wg_mask=0, data_size=2B
  g1[1] = 0;                                         // abar=0; tensor_dim0 lo16=0
  g1[2] = (int)0x4000;                               // tensor_dim0 = 1<<30
  g1[3] = (int)((tile_x << 16) | 0x4000u);           // tensor_dim1 = 1<<30; tile_dim0
  g1[4] = (int)tile_y;                               // tile_dim1; tile_dim2=0
  g1[5] = (int)stride_x;                             // tensor_dim0_stride lo32
  g1[6] = 0;                                         // stride hi; dim1_stride lo16
  g1[7] = 0x10;                                      // tensor_dim1_stride = 1<<20
  i32x4 z4;
  z4[0] = 0; z4[1] = 0; z4[2] = 0; z4[3] = 0;
  i32x8 z8;
#pragma unroll
  for (int i = 0; i < 8; ++i) z8[i] = 0;
  __builtin_amdgcn_tensor_load_to_lds(g0, g1, z4, z4, z8, 0);
}
#endif

// ---------------- Kernel 1a: x fp32 -> bf16 ----------------
__global__ void __launch_bounds__(256) cvt_x(const float* __restrict__ x,
                                             __bf16* __restrict__ xb) {
  size_t i = ((size_t)blockIdx.x * 256 + threadIdx.x) * 8;
  float4 a = *(const float4*)(x + i);
  float4 b = *(const float4*)(x + i + 4);
  v8bf o;
  o[0] = f2bf(a.x); o[1] = f2bf(a.y); o[2] = f2bf(a.z); o[3] = f2bf(a.w);
  o[4] = f2bf(b.x); o[5] = f2bf(b.y); o[6] = f2bf(b.z); o[7] = f2bf(b.w);
  *(v8bf*)(xb + i) = o;
}

// ---------------- Kernel 1b: W fp32 [K][N] -> Wt bf16 [3][N][K] ----------------
__global__ void __launch_bounds__(256) cvt_w(const float* __restrict__ Wq,
                                             const float* __restrict__ Wk,
                                             const float* __restrict__ Wv,
                                             __bf16* __restrict__ Wt) {
  unsigned tid = blockIdx.x * 256 + threadIdx.x;   // 0 .. 3*1024*1024-1
  unsigned n = tid & 1023u;
  unsigned k = (tid >> 10) & 1023u;
  unsigned m = tid >> 20;
  const float* W = (m == 0) ? Wq : ((m == 1) ? Wk : Wv);
  Wt[((size_t)m << 20) + ((size_t)n << 10) + k] = f2bf(W[((size_t)k << 10) + n]);
}

// ---------------- Kernel 2: fused QKV projection GEMM ----------------
// Y[m] = Xb(16384x1024) * Wt[m]^T ; Q,K row-major bf16; V stored transposed.
// W tile (128x32 bf16) staged in LDS; TDM double-buffered when available.
__global__ void __launch_bounds__(256) gemm_qkv(const __bf16* __restrict__ Xb,
                                                const __bf16* __restrict__ Wt,
                                                __bf16* __restrict__ Qb,
                                                __bf16* __restrict__ Kb,
                                                __bf16* __restrict__ Vt) {
  __shared__ __bf16 lw[2][128 * 32];               // 2 x 8 KB W tiles
  const int mat = blockIdx.z;
  const int n0 = blockIdx.x * 128;
  const int m0 = blockIdx.y * 128;
  const int t = threadIdx.x;
  const int w = t >> 5;
  const int lane = t & 31, ln = lane & 15, half = lane >> 4;
  const int mw = m0 + w * 16;
  const __bf16* Wm = Wt + ((size_t)mat << 20);
  const __bf16* wtile0 = Wm + ((size_t)n0 << 10);

  v8f acc[8];
#pragma unroll
  for (int s = 0; s < 8; ++s)
#pragma unroll
    for (int i = 0; i < 8; ++i) acc[s][i] = 0.0f;

  const __bf16* arow = Xb + ((size_t)(mw + ln) << 10) + half * 8;

#ifdef USE_TDM
  // ---- TDM double-buffered pipeline: wave 0 drives the DMA ----
  __bf16* lwp = &lw[0][0];
  if (w == 0) tdm_load_2d(0u, wtile0, 32, 128, 1024);
  for (int kt = 0; kt < 32; ++kt) {
    const int k0 = kt * 32;
    if (w == 0) {
      if (kt + 1 < 32) {
        tdm_load_2d((unsigned)(((kt + 1) & 1) * 8192),
                    wtile0 + (kt + 1) * 32, 32, 128, 1024);
        __builtin_amdgcn_s_wait_tensorcnt(1);      // drain load(kt), keep kt+1
      } else {
        __builtin_amdgcn_s_wait_tensorcnt(0);      // last tile: drain all
      }
    }
    __syncthreads();                               // buffer kt&1 is ready
    // The TDM engine wrote lw behind the compiler's back: escape the pointer
    // and clobber memory so the B-fragment ds_loads are real and not hoisted.
    asm volatile("" : : "r"(lwp) : "memory");
    const __bf16* lwb = &lw[kt & 1][0];
    v16bf a = ld_a16(arow + k0, arow + k0 + 16);
#pragma unroll
    for (int sub = 0; sub < 8; ++sub) {
      v16bf b = ld_a16(&lwb[(sub * 16 + ln) * 32 + half * 8], &lwb[(sub * 16 + ln) * 32 + half * 8 + 16]);
      acc[sub] = wmma_bf16(a, b, acc[sub]);
    }
    __syncthreads();                               // done reading before reuse
  }
#else
  // ---- fallback: cooperative synchronous staging ----
  for (int k0 = 0; k0 < 1024; k0 += 32) {
#pragma unroll
    for (int it = 0; it < 2; ++it) {
      int idx = t + it * 256;                      // 0..511
      int row = idx >> 2, seg = idx & 3;
      *(v8bf*)&lw[0][row * 32 + seg * 8] =
          *(const v8bf*)(wtile0 + ((size_t)row << 10) + k0 + seg * 8);
    }
    __syncthreads();
    v16bf a = ld_a16(arow + k0, arow + k0 + 16);
#pragma unroll
    for (int sub = 0; sub < 8; ++sub) {
      v16bf b = ld_a16(&lw[0][(sub * 16 + ln) * 32 + half * 8], &lw[0][(sub * 16 + ln) * 32 + half * 8 + 16]);
      acc[sub] = wmma_bf16(a, b, acc[sub]);
    }
    __syncthreads();
  }
#endif

#pragma unroll
  for (int sub = 0; sub < 8; ++sub) {
    const int col = n0 + sub * 16 + ln;
#pragma unroll
    for (int r = 0; r < 8; ++r) {
      const int rowg = mw + r + half * 8;
      __bf16 v = f2bf(acc[sub][r]);
      if (mat == 0) {
        Qb[((size_t)rowg << 10) + col] = v;
      } else if (mat == 1) {
        Kb[((size_t)rowg << 10) + col] = v;
      } else {
        const int bb = rowg >> 12, srow = rowg & 4095;
        Vt[((size_t)bb << 22) + ((size_t)col << 12) + srow] = v;   // Vt[b][d][s]
      }
    }
  }
}

// ---------------- Kernel 3: flash attention (causal) ----------------
// Block: (qb, dslice, batch). 8 waves, wave owns 16 query rows, Dv=256 slice.
__global__ void __launch_bounds__(256) attn(const __bf16* __restrict__ Qb,
                                            const __bf16* __restrict__ Kb,
                                            const __bf16* __restrict__ Vt,
                                            float* __restrict__ out) {
  __shared__ __bf16 plds[8 * 1024];                // per-wave 16x64 P tile
  const int bb = blockIdx.z;
  const int v0 = blockIdx.y * 256;
  const int w = threadIdx.x >> 5;
  const int q0 = blockIdx.x * 128 + w * 16;
  const int lane = threadIdx.x & 31, ln = lane & 15, half = lane >> 4;
  const int half8 = half * 8;
  __bf16* pw = &plds[w * 1024];

  const __bf16* qrow = Qb + ((size_t)(bb * 4096 + q0 + ln) << 10) + half8;
  const __bf16* kbase = Kb + ((size_t)bb << 22);
  const __bf16* vbase = Vt + ((size_t)bb << 22);

  v8f o[16];
#pragma unroll
  for (int d = 0; d < 16; ++d)
#pragma unroll
    for (int i = 0; i < 8; ++i) o[d][i] = 0.0f;

  float m_st[8], l_st[8], corr[8];
#pragma unroll
  for (int r = 0; r < 8; ++r) { m_st[r] = -1e30f; l_st[r] = 0.0f; }

  const float sc = 0.04508422002778f;              // log2(e) / sqrt(1024)
  const int qmax = q0 + 15;

  for (int key0 = 0; key0 <= qmax; key0 += 64) {
    if (key0 + 64 < 4096)                          // warm L2/L0 for next K tile
      __builtin_prefetch(kbase + ((size_t)(key0 + 64 + ln) << 10), 0, 0);

    // ---- S = Q K^T over full D=1024 ----
    v8f s4[4];
#pragma unroll
    for (int s = 0; s < 4; ++s)
#pragma unroll
      for (int i = 0; i < 8; ++i) s4[s][i] = 0.0f;

    for (int d0 = 0; d0 < 1024; d0 += 32) {
      v16bf a = ld_a16(qrow + d0, qrow + d0 + 16);
#pragma unroll
      for (int sub = 0; sub < 4; ++sub) {
        const __bf16* kp =
            kbase + ((size_t)(key0 + sub * 16 + ln) << 10) + d0 + half * 8;
        s4[sub] = wmma_bf16(a, ld_a16(kp, kp + 16), s4[sub]);
      }
    }

    // ---- scale + causal mask ----
    const bool full = (key0 + 63 <= q0);
#pragma unroll
    for (int sub = 0; sub < 4; ++sub) {
      const int kcol = key0 + sub * 16 + ln;
#pragma unroll
      for (int r = 0; r < 8; ++r) {
        float v = s4[sub][r] * sc;
        s4[sub][r] = (full || kcol <= q0 + r + half8) ? v : -1e30f;
      }
    }

    // ---- online softmax (row stats via 16-lane xor reductions) ----
#pragma unroll
    for (int r = 0; r < 8; ++r) {
      float mx = fmaxf(fmaxf(s4[0][r], s4[1][r]), fmaxf(s4[2][r], s4[3][r]));
      mx = fmaxf(mx, __shfl_xor(mx, 1));
      mx = fmaxf(mx, __shfl_xor(mx, 2));
      mx = fmaxf(mx, __shfl_xor(mx, 4));
      mx = fmaxf(mx, __shfl_xor(mx, 8));
      float mn = fmaxf(m_st[r], mx);
      float c = EXP2F(m_st[r] - mn);
      m_st[r] = mn;
      corr[r] = c;
      float rs = 0.0f;
#pragma unroll
      for (int sub = 0; sub < 4; ++sub) {
        float p = EXP2F(s4[sub][r] - mn);
        s4[sub][r] = p;
        rs += p;
      }
      rs += __shfl_xor(rs, 1);
      rs += __shfl_xor(rs, 2);
      rs += __shfl_xor(rs, 4);
      rs += __shfl_xor(rs, 8);
      l_st[r] = l_st[r] * c + rs;
    }

    // ---- rescale O ----
#pragma unroll
    for (int dsub = 0; dsub < 16; ++dsub)
#pragma unroll
      for (int r = 0; r < 8; ++r) o[dsub][r] *= corr[r];

    // ---- P -> LDS (C-layout -> row-major bf16), then O += P V ----
#pragma unroll
    for (int sub = 0; sub < 4; ++sub)
#pragma unroll
      for (int r = 0; r < 8; ++r)
        pw[(r + half8) * 64 + sub * 16 + ln] = f2bf(s4[sub][r]);

#pragma unroll
    for (int kk = 0; kk < 2; ++kk) {
      const __bf16* pb = &pw[ln * 64 + kk * 32 + half8];
      v16bf ap = ld_a16(pb, pb + 16);
#pragma unroll
      for (int dsub = 0; dsub < 16; ++dsub) {
        const __bf16* vp = vbase + ((size_t)(v0 + dsub * 16 + ln) << 12) +
                           key0 + kk * 32 + half * 8;
        o[dsub] = wmma_bf16(ap, ld_a16(vp, vp + 16), o[dsub]);
      }
    }
  }

  // ---- normalize and store fp32 ----
#pragma unroll
  for (int r = 0; r < 8; ++r) corr[r] = 1.0f / l_st[r];
#pragma unroll
  for (int dsub = 0; dsub < 16; ++dsub)
#pragma unroll
    for (int r = 0; r < 8; ++r)
      out[((size_t)(bb * 4096 + q0 + r + half8) << 10) + v0 + dsub * 16 + ln] =
          o[dsub][r] * corr[r];
}

extern "C" void kernel_launch(void* const* d_in, const int* in_sizes, int n_in,
                              void* d_out, int out_size, void* d_ws, size_t ws_size,
                              hipStream_t stream) {
  const float* x  = (const float*)d_in[0];
  const float* Wq = (const float*)d_in[1];
  const float* Wk = (const float*)d_in[2];
  const float* Wv = (const float*)d_in[3];
  float* out = (float*)d_out;

  char* ws = (char*)d_ws;
  __bf16* Xb = (__bf16*)(ws);                       // 16384*1024*2   = 32 MB
  __bf16* Wt = (__bf16*)(ws + 33554432);            // 3*1024*1024*2  = 6 MB
  __bf16* Qb = (__bf16*)(ws + 39845888);            // 32 MB
  __bf16* Kb = (__bf16*)(ws + 73400320);            // 32 MB
  __bf16* Vt = (__bf16*)(ws + 106954752);           // 32 MB (ends 140509184)

  cvt_x<<<8192, 256, 0, stream>>>(x, Xb);
  cvt_w<<<12288, 256, 0, stream>>>(Wq, Wk, Wv, Wt);
  gemm_qkv<<<dim3(8, 128, 3), 256, 0, stream>>>(Xb, Wt, Qb, Kb, Vt);
  attn<<<dim3(32, 4, 4), 256, 0, stream>>>(Qb, Kb, Vt, out);
}
